// GNNGWLoss_43654047596884
// MI455X (gfx1250) — compile-verified, matrix operand order checked by reading, not run on hardware
//
#include <hip/hip_runtime.h>
#include <hip/hip_bf16.h>
#include <stdint.h>

typedef uint16_t u16;
typedef uint32_t u32;
typedef __attribute__((ext_vector_type(16))) __bf16 v16bf;
typedef __attribute__((ext_vector_type(8)))  float  v8f;

#define FEAT 128
#define NROW 4096
#define KNN  8
#define NCLS 100
#define GAMMA 1.0f
#define SINK_ITERS 20

// ---------------- helpers ----------------

__device__ __forceinline__ u16 f2bf(float x) {
  u32 u = __float_as_uint(x);
  u32 r = (u + 0x7FFFu + ((u >> 16) & 1u)) >> 16;
  return (u16)r;
}

union FragU { v16bf v; uint4 q[2]; };

// Load a 16x32 bf16 WMMA A/B fragment from a row-major matrix.
// ISA 16-bit A layout: lane h=lane>>4, m=lane&15; VGPR0..3 = K 8h..8h+7,
// VGPR4..7 = K 16+8h..16+8h+7 -> two contiguous 16B loads per lane.
__device__ __forceinline__ v16bf load_frag(const u16* __restrict__ p, int ld, int lane) {
  int h = (lane >> 4) & 1;
  int m = lane & 15;
  const u16* row = p + (size_t)m * ld + 8 * h;
  FragU f;
  f.q[0] = *reinterpret_cast<const uint4*>(row);
  f.q[1] = *reinterpret_cast<const uint4*>(row + 16);
  return f.v;
}

// Async-stage a float vector from global into LDS (global -> LDS directly,
// no VGPR round trip), then sync the block.
__device__ __forceinline__ void async_stage_vec(const float* __restrict__ g,
                                                float* lds, int n, int tid, int nthreads) {
  for (int j = tid * 4; j < n; j += nthreads * 4) {
    u32 ldsoff = (u32)(uintptr_t)(lds + j);     // LDS aperture: addr[31:0] = LDS offset
    const float* ga = g + j;
    asm volatile("global_load_async_to_lds_b128 %0, %1, off"
                 :: "v"(ldsoff), "v"(ga) : "memory");
  }
  asm volatile("s_wait_asynccnt 0x0" ::: "memory");
  __syncthreads();
}

// ---------------- GEMM: C(MxN,f32) = A(MxK,bf16) * Bt(NxK,bf16)^T ------------
// One 16 x (16*TN) strip per wave32: A fragment reused TN times, software
// pipelined so next-k fragments load while current WMMAs execute.
// Split-K over gridDim.y: block row y covers K range [y*Kc, (y+1)*Kc) and
// writes its partial result to C + y*M*N.
template <int TN>
__global__ void gemm_bf16_wmma(const u16* __restrict__ A, const u16* __restrict__ Bt,
                               float* __restrict__ C, int M, int N, int K, int Kc) {
  int wave = (int)((blockIdx.x * blockDim.x + threadIdx.x) >> 5);
  int lane = threadIdx.x & 31;
  int groupsN = N / (16 * TN);
  int strips  = (M >> 4) * groupsN;
  if (wave >= strips) return;
  int tm = wave / groupsN, tg = wave % groupsN;
  int kOff = blockIdx.y * Kc;
  const u16* a0 = A  + (size_t)tm * 16 * K + kOff;
  const u16* b0 = Bt + (size_t)tg * (16 * TN) * K + kOff;
  int h = (lane >> 4) & 1, m = lane & 15;

  v8f acc[TN];
  #pragma unroll
  for (int t = 0; t < TN; ++t) acc[t] = (v8f){};

  v16bf a_cur = load_frag(a0, K, lane);
  v16bf b_cur[TN];
  #pragma unroll
  for (int t = 0; t < TN; ++t) b_cur[t] = load_frag(b0 + (size_t)t * 16 * K, K, lane);

  for (int k = 32; k < Kc; k += 32) {
    // prefetch ahead (speculative; dropped if OOB)
    __builtin_prefetch(a0 + (size_t)m * K + k + 96, 0, 3);
    // issue next-step loads before consuming current fragments
    v16bf a_nxt = load_frag(a0 + k, K, lane);
    v16bf b_nxt[TN];
    #pragma unroll
    for (int t = 0; t < TN; ++t) b_nxt[t] = load_frag(b0 + (size_t)t * 16 * K + k, K, lane);
    #pragma unroll
    for (int t = 0; t < TN; ++t)
      acc[t] = __builtin_amdgcn_wmma_f32_16x16x32_bf16(false, a_cur, false, b_cur[t],
                                                       (short)0, acc[t], false, false);
    a_cur = a_nxt;
    #pragma unroll
    for (int t = 0; t < TN; ++t) b_cur[t] = b_nxt[t];
  }
  #pragma unroll
  for (int t = 0; t < TN; ++t)
    acc[t] = __builtin_amdgcn_wmma_f32_16x16x32_bf16(false, a_cur, false, b_cur[t],
                                                     (short)0, acc[t], false, false);

  float* Cs = C + (size_t)blockIdx.y * M * N;
  #pragma unroll
  for (int t = 0; t < TN; ++t) {
    int tn = tg * TN + t;
    float* c0 = Cs + (size_t)(tm * 16) * N + tn * 16;
    #pragma unroll
    for (int r = 0; r < 8; ++r)
      c0[(size_t)(r + 8 * h) * N + m] = acc[t][r];
  }
}

// sum split-K partials + bias (N must be power of two)
__global__ void reduce_bias(const float* __restrict__ part, const float* __restrict__ bias,
                            float* __restrict__ out, int MN, int N, int splits) {
  int i = blockIdx.x * 256 + threadIdx.x;
  if (i >= MN) return;
  float s = 0.0f;
  for (int t = 0; t < splits; ++t) s += part[(size_t)t * MN + i];
  out[i] = s + bias[i & (N - 1)];
}

// ---------------- elementwise converts ----------------

__global__ void cvt_f32_bf16(const float* __restrict__ in, u16* __restrict__ out, long n) {
  long i = (long)blockIdx.x * blockDim.x + threadIdx.x;
  long stride = (long)gridDim.x * blockDim.x;
  for (; i < n; i += stride) out[i] = f2bf(in[i]);
}

// in: K x N row-major (f32) -> out: N x K row-major (bf16)
__global__ void transpose_cvt(const float* __restrict__ in, u16* __restrict__ out,
                              int K, int N) {
  long i = (long)blockIdx.x * blockDim.x + threadIdx.x;
  long tot = (long)K * N;
  if (i >= tot) return;
  int nIdx = (int)(i / K), kIdx = (int)(i % K);
  out[i] = f2bf(in[(size_t)kIdx * N + nIdx]);
}

// ---------------- row l2norm (FEAT=128), emit f32 + bf16 ----------------

__global__ void l2norm_rows(const float* __restrict__ Z, float* __restrict__ outF,
                            u16* __restrict__ outB) {
  int row = blockIdx.x, tid = threadIdx.x; // 128 threads
  float v = Z[(size_t)row * FEAT + tid];
  __shared__ float red[FEAT];
  red[tid] = v * v; __syncthreads();
  for (int s = 64; s > 0; s >>= 1) { if (tid < s) red[tid] += red[tid + s]; __syncthreads(); }
  float o = v * (1.0f / sqrtf(red[0]));
  outF[(size_t)row * FEAT + tid] = o;
  outB[(size_t)row * FEAT + tid] = f2bf(o);
}

// ---------------- softmax rows + row L2 norm of softmax ----------------

__global__ void softmax_norm(const float* __restrict__ L, float* __restrict__ S,
                             float* __restrict__ Wn) {
  int row = blockIdx.x, tid = threadIdx.x; // 128 threads, NCLS=100 active
  __shared__ float red[128];
  float x = (tid < NCLS) ? L[(size_t)row * NCLS + tid] : -3.4e38f;
  red[tid] = x; __syncthreads();
  for (int s = 64; s > 0; s >>= 1) { if (tid < s) red[tid] = fmaxf(red[tid], red[tid + s]); __syncthreads(); }
  float mx = red[0]; __syncthreads();
  float e = (tid < NCLS) ? expf(x - mx) : 0.0f;
  red[tid] = e; __syncthreads();
  for (int s = 64; s > 0; s >>= 1) { if (tid < s) red[tid] += red[tid + s]; __syncthreads(); }
  float ssum = red[0]; __syncthreads();
  float sv = (tid < NCLS) ? e / ssum : 0.0f;
  if (tid < NCLS) S[(size_t)row * NCLS + tid] = sv;
  red[tid] = sv * sv; __syncthreads();
  for (int s = 64; s > 0; s >>= 1) { if (tid < s) red[tid] += red[tid + s]; __syncthreads(); }
  if (tid == 0) Wn[row] = sqrtf(red[0]);
}

// ---------------- kNN (cosine distance on softmax rows, top-8 smallest) ----

__global__ void knn_topk(const float* __restrict__ S, const float* __restrict__ Wn,
                         int* __restrict__ idx) {
  int row = blockIdx.x, tid = threadIdx.x; // 256 threads
  __shared__ float si[128];
  if (tid < NCLS) si[tid] = S[(size_t)row * NCLS + tid];
  __syncthreads();
  float wi = Wn[row];
  float bd[KNN]; int bi[KNN];
  #pragma unroll
  for (int t = 0; t < KNN; ++t) { bd[t] = 3.4e38f; bi[t] = -1; }
  for (int j = tid; j < NROW; j += 256) {
    const float* sj = S + (size_t)j * NCLS;
    float dot = 0.0f;
    for (int c = 0; c < NCLS; ++c) dot += si[c] * sj[c];
    float den = fmaxf(wi * Wn[j], 1e-7f);
    float d = 1.0f - dot / den;
    if (j == row) d = -1.0f; // dist*(1-I) - I
    if (d < bd[KNN - 1]) {
      int p = KNN - 1;
      while (p > 0 && bd[p - 1] > d) { bd[p] = bd[p - 1]; bi[p] = bi[p - 1]; --p; }
      bd[p] = d; bi[p] = j;
    }
  }
  __shared__ float cd[256 * KNN];
  __shared__ int   ci[256 * KNN];
  #pragma unroll
  for (int t = 0; t < KNN; ++t) { cd[tid * KNN + t] = bd[t]; ci[tid * KNN + t] = bi[t]; }
  __syncthreads();
  __shared__ float rd[256]; __shared__ int ri[256];
  for (int t = 0; t < KNN; ++t) {
    float mn = 3.4e38f; int mi = tid * KNN;
    #pragma unroll
    for (int q = 0; q < KNN; ++q) {
      float d = cd[tid * KNN + q];
      if (d < mn) { mn = d; mi = tid * KNN + q; }
    }
    rd[tid] = mn; ri[tid] = mi; __syncthreads();
    for (int s = 128; s > 0; s >>= 1) {
      if (tid < s && rd[tid + s] < rd[tid]) { rd[tid] = rd[tid + s]; ri[tid] = ri[tid + s]; }
      __syncthreads();
    }
    if (tid == 0) { idx[row * KNN + t] = ci[ri[0]]; cd[ri[0]] = 3.4e38f; }
    __syncthreads();
  }
}

// ---------------- build [h, mean(h[idx])] as bf16 (NROW x 2*FEAT) -----------

__global__ void build_hcat(const float* __restrict__ h, const int* __restrict__ idx,
                           u16* __restrict__ hcat) {
  long i = (long)blockIdx.x * blockDim.x + threadIdx.x;
  if (i >= (long)NROW * 2 * FEAT) return;
  int row = (int)(i >> 8), c = (int)(i & 255);
  float v;
  if (c < FEAT) v = h[(size_t)row * FEAT + c];
  else {
    float s = 0.0f;
    #pragma unroll
    for (int t = 0; t < KNN; ++t) s += h[(size_t)idx[row * KNN + t] * FEAT + (c - FEAT)];
    v = s * (1.0f / KNN);
  }
  hcat[i] = f2bf(v);
}

// ---------------- global min/max over M ----------------

__global__ void minmax_part(const float* __restrict__ X, float* __restrict__ pmin,
                            float* __restrict__ pmax, long n) {
  long i = (long)blockIdx.x * 256 + threadIdx.x;
  long stride = (long)gridDim.x * 256;
  float mn = 3.4e38f, mx = -3.4e38f;
  for (; i < n; i += stride) { float x = X[i]; mn = fminf(mn, x); mx = fmaxf(mx, x); }
  __shared__ float smn[256], smx[256];
  smn[threadIdx.x] = mn; smx[threadIdx.x] = mx; __syncthreads();
  for (int s = 128; s > 0; s >>= 1) {
    if (threadIdx.x < s) {
      smn[threadIdx.x] = fminf(smn[threadIdx.x], smn[threadIdx.x + s]);
      smx[threadIdx.x] = fmaxf(smx[threadIdx.x], smx[threadIdx.x + s]);
    }
    __syncthreads();
  }
  if (threadIdx.x == 0) { pmin[blockIdx.x] = smn[0]; pmax[blockIdx.x] = smx[0]; }
}

__global__ void minmax_final(const float* __restrict__ pmin, const float* __restrict__ pmax,
                             float* __restrict__ gmm, int nb) {
  int tid = threadIdx.x;
  float mn = 3.4e38f, mx = -3.4e38f;
  for (int i = tid; i < nb; i += 256) { mn = fminf(mn, pmin[i]); mx = fmaxf(mx, pmax[i]); }
  __shared__ float smn[256], smx[256];
  smn[tid] = mn; smx[tid] = mx; __syncthreads();
  for (int s = 128; s > 0; s >>= 1) {
    if (tid < s) { smn[tid] = fminf(smn[tid], smn[tid + s]); smx[tid] = fmaxf(smx[tid], smx[tid + s]); }
    __syncthreads();
  }
  if (tid == 0) { gmm[0] = smn[0]; gmm[1] = smx[0]; }
}

// P0[i][j] = exp(-gamma*(rowmax(M_i) - M_ij)/range); also init v=1, opt. write Mn
__global__ void build_P0(const float* __restrict__ Mraw, const float* __restrict__ gmm,
                         float* __restrict__ P0, float* __restrict__ v,
                         float* __restrict__ Mout, int n) {
  int row = blockIdx.x, tid = threadIdx.x; // 256 threads
  __shared__ float red[256];
  const float* mrow = Mraw + (size_t)row * n;
  float mx = -3.4e38f;
  for (int j = tid; j < n; j += 256) mx = fmaxf(mx, mrow[j]);
  red[tid] = mx; __syncthreads();
  for (int s = 128; s > 0; s >>= 1) { if (tid < s) red[tid] = fmaxf(red[tid], red[tid + s]); __syncthreads(); }
  float rowmax = red[0];
  float gmin = gmm[0], invr = 1.0f / (gmm[1] - gmm[0]);
  for (int j = tid; j < n; j += 256) {
    float m = mrow[j];
    P0[(size_t)row * n + j] = expf(-GAMMA * (rowmax - m) * invr);
    if (Mout) Mout[(size_t)row * n + j] = (m - gmin) * invr;
  }
  if (tid == 0) v[row] = 1.0f;
}

// u_i = r / sum_j P0[i][j] v[j]
__global__ void sinkhorn_row(const float* __restrict__ P0, const float* __restrict__ v,
                             float* __restrict__ u, int n, float r) {
  int row = blockIdx.x, tid = threadIdx.x; // 256
  __shared__ float vs[NROW];
  __shared__ float red[256];
  async_stage_vec(v, vs, n, tid, 256);
  const float* p = P0 + (size_t)row * n;
  float s = 0.0f;
  for (int j = tid; j < n; j += 256) s += p[j] * vs[j];
  red[tid] = s; __syncthreads();
  for (int st = 128; st > 0; st >>= 1) { if (tid < st) red[tid] += red[tid + st]; __syncthreads(); }
  if (tid == 0) u[row] = r / red[0];
}

// v_j = c / sum_i P0[i][j] u[i]
__global__ void sinkhorn_col(const float* __restrict__ P0, const float* __restrict__ u,
                             float* __restrict__ v, int n, float c) {
  int j = blockIdx.x * 256 + threadIdx.x;
  __shared__ float us[NROW];
  async_stage_vec(u, us, n, threadIdx.x, 256);
  if (j >= n) return;
  float s = 0.0f;
  for (int i = 0; i < n; ++i) s += P0[(size_t)i * n + j] * us[i];
  v[j] = c / s;
}

// final row-normalize; sum-of-squares of (P - I) per row; optional P output
__global__ void sinkhorn_final(const float* __restrict__ P0, const float* __restrict__ v,
                               float* __restrict__ Pout, float* __restrict__ rowSq, int n) {
  int row = blockIdx.x, tid = threadIdx.x; // 256
  __shared__ float vs[NROW];
  __shared__ float red[256];
  async_stage_vec(v, vs, n, tid, 256);
  const float* p = P0 + (size_t)row * n;
  float t = 0.0f;
  for (int j = tid; j < n; j += 256) t += p[j] * vs[j];
  red[tid] = t; __syncthreads();
  for (int s = 128; s > 0; s >>= 1) { if (tid < s) red[tid] += red[tid + s]; __syncthreads(); }
  float inv = 1.0f / red[0];
  __syncthreads();
  float sq = 0.0f;
  for (int j = tid; j < n; j += 256) {
    float pf = p[j] * vs[j] * inv;
    if (Pout) Pout[(size_t)row * n + j] = pf;
    float d = pf - ((j == row) ? 1.0f : 0.0f);
    sq += d * d;
  }
  red[tid] = sq; __syncthreads();
  for (int s = 128; s > 0; s >>= 1) { if (tid < s) red[tid] += red[tid + s]; __syncthreads(); }
  if (tid == 0) rowSq[row] = red[0];
}

__global__ void loss_finish(const float* __restrict__ rowSq, float* __restrict__ out, int n) {
  int tid = threadIdx.x;
  __shared__ float red[256];
  float s = 0.0f;
  for (int i = tid; i < n; i += 256) s += rowSq[i];
  red[tid] = s; __syncthreads();
  for (int st = 128; st > 0; st >>= 1) { if (tid < st) red[tid] += red[tid + st]; __syncthreads(); }
  if (tid == 0) out[0] = sqrtf(red[0]);
}

__global__ void total_loss(float* __restrict__ out) {
  if (threadIdx.x == 0) out[0] = out[1] + out[2] + out[3] + out[4];
}

// ---------------- host-side orchestration ----------------

#define TNB 4      // 16x64 strip per wave
#define KSPLIT 4   // split-K for the K-deep GEMMs

// small GEMM + bias + l2norm pipeline: out = l2norm(A @ Bt^T + bias)
static void run_linear(const u16* A, const u16* Bt, const float* bias,
                       float* cpart, float* ztmp, float* outF, u16* outB,
                       int K, hipStream_t stream) {
  int waves = (NROW / 16) * (FEAT / (16 * TNB));
  dim3 grid(waves / 8, KSPLIT);
  gemm_bf16_wmma<TNB><<<grid, 256, 0, stream>>>(A, Bt, cpart, NROW, FEAT, K, K / KSPLIT);
  reduce_bias<<<(NROW * FEAT) / 256, 256, 0, stream>>>(cpart, bias, ztmp, NROW * FEAT, FEAT, KSPLIT);
  l2norm_rows<<<NROW, 128, 0, stream>>>(ztmp, outF, outB);
}

static void run_ot_loss(const u16* ftb, const u16* fsb,
                        float* Mbuf, float* P0buf, float* u, float* v,
                        float* pmin, float* pmax, float* gmm, float* rowSq,
                        float* loss_out, float* P_out, float* M_out,
                        hipStream_t stream) {
  const int n = NROW;
  // M = ft @ fs^T  (fsb row-major == B^T row-major); single K chunk (K=128)
  int waves = (n / 16) * (n / (16 * TNB));
  dim3 grid(waves / 8, 1);
  gemm_bf16_wmma<TNB><<<grid, 256, 0, stream>>>(ftb, fsb, Mbuf, n, n, FEAT, FEAT);
  minmax_part<<<1024, 256, 0, stream>>>(Mbuf, pmin, pmax, (long)n * n);
  minmax_final<<<1, 256, 0, stream>>>(pmin, pmax, gmm, 1024);
  build_P0<<<n, 256, 0, stream>>>(Mbuf, gmm, P0buf, v, M_out, n);
  const float rc = 1.0f / n;
  for (int it = 0; it < SINK_ITERS; ++it) {
    sinkhorn_row<<<n, 256, 0, stream>>>(P0buf, v, u, n, rc);
    sinkhorn_col<<<n / 256, 256, 0, stream>>>(P0buf, u, v, n, rc);
  }
  sinkhorn_final<<<n, 256, 0, stream>>>(P0buf, v, P_out, rowSq, n);
  loss_finish<<<1, 256, 0, stream>>>(rowSq, loss_out, n);
}

extern "C" void kernel_launch(void* const* d_in, const int* in_sizes, int n_in,
                              void* d_out, int out_size, void* d_ws, size_t ws_size,
                              hipStream_t stream) {
  const float* f_s  = (const float*)d_in[1];
  const float* l_s  = (const float*)d_in[2];
  const float* f_t  = (const float*)d_in[3];
  const float* l_t  = (const float*)d_in[4];
  const float* W_es = (const float*)d_in[5];
  const float* b_es = (const float*)d_in[6];
  const float* W_et = (const float*)d_in[7];
  const float* b_et = (const float*)d_in[8];
  const float* W_gs = (const float*)d_in[9];
  const float* b_gs = (const float*)d_in[10];
  const float* W_gt = (const float*)d_in[11];
  const float* b_gt = (const float*)d_in[12];
  float* out = (float*)d_out;

  char* ws = (char*)d_ws;
  size_t cur = 0;
  auto take = [&](size_t bytes) -> void* {
    void* p = ws + cur;
    cur += (bytes + 255) & ~(size_t)255;
    return p;
  };

  const long NFS = (long)NROW * 8192;
  u16* fsb  = (u16*)take(NFS * 2);            // reused as Mbuf after embeds
  u16* ftb  = (u16*)take(NFS * 2);            // reused as P0buf after embeds
  u16* wtes = (u16*)take((size_t)FEAT * 8192 * 2);
  u16* wtet = (u16*)take((size_t)FEAT * 8192 * 2);
  u16* wtgs = (u16*)take((size_t)FEAT * 2 * FEAT * 2);
  u16* wtgt = (u16*)take((size_t)FEAT * 2 * FEAT * 2);
  float* fes_f = (float*)take((size_t)NROW * FEAT * 4);
  float* fet_f = (float*)take((size_t)NROW * FEAT * 4);
  float* fgs_f = (float*)take((size_t)NROW * FEAT * 4);
  float* fgt_f = (float*)take((size_t)NROW * FEAT * 4);
  u16* fes_b = (u16*)take((size_t)NROW * FEAT * 2);
  u16* fet_b = (u16*)take((size_t)NROW * FEAT * 2);
  u16* fgs_b = (u16*)take((size_t)NROW * FEAT * 2);
  u16* fgt_b = (u16*)take((size_t)NROW * FEAT * 2);
  float* cpart = (float*)take((size_t)KSPLIT * NROW * FEAT * 4);
  float* ztmp  = (float*)take((size_t)NROW * FEAT * 4);
  float* Ss  = (float*)take((size_t)NROW * NCLS * 4);
  float* St  = (float*)take((size_t)NROW * NCLS * 4);
  float* wns = (float*)take((size_t)NROW * 4);
  float* wnt = (float*)take((size_t)NROW * 4);
  int*   idxs = (int*)take((size_t)NROW * KNN * 4);
  int*   idxt = (int*)take((size_t)NROW * KNN * 4);
  u16* hcs = (u16*)take((size_t)NROW * 2 * FEAT * 2);
  u16* hct = (u16*)take((size_t)NROW * 2 * FEAT * 2);
  float* u    = (float*)take((size_t)NROW * 4);
  float* v    = (float*)take((size_t)NROW * 4);
  float* pmin = (float*)take(1024 * 4);
  float* pmax = (float*)take(1024 * 4);
  float* gmm  = (float*)take(256);
  float* rowSq = (float*)take((size_t)NROW * 4);

  float* Mbuf  = (float*)fsb;  // 4096*4096*4 == 4096*8192*2
  float* P0buf = (float*)ftb;

  // --- convert inputs to bf16 (weights transposed) ---
  cvt_f32_bf16<<<4096, 256, 0, stream>>>(f_s, fsb, NFS);
  cvt_f32_bf16<<<4096, 256, 0, stream>>>(f_t, ftb, NFS);
  transpose_cvt<<<(8192 * FEAT + 255) / 256, 256, 0, stream>>>(W_es, wtes, 8192, FEAT);
  transpose_cvt<<<(8192 * FEAT + 255) / 256, 256, 0, stream>>>(W_et, wtet, 8192, FEAT);
  transpose_cvt<<<(2 * FEAT * FEAT + 255) / 256, 256, 0, stream>>>(W_gs, wtgs, 2 * FEAT, FEAT);
  transpose_cvt<<<(2 * FEAT * FEAT + 255) / 256, 256, 0, stream>>>(W_gt, wtgt, 2 * FEAT, FEAT);

  // --- embeds: f_e* = l2norm(f @ W + b) (split-K=4 for grid utilization) ---
  run_linear(fsb, wtes, b_es, cpart, ztmp, fes_f, fes_b, 8192, stream);
  run_linear(ftb, wtet, b_et, cpart, ztmp, fet_f, fet_b, 8192, stream);
  // (fsb/ftb dead from here on -> reused as Mbuf/P0buf)

  // --- loss_e = ot(f_et, f_es) ---
  run_ot_loss(fet_b, fes_b, Mbuf, P0buf, u, v, pmin, pmax, gmm, rowSq,
              out + 1, nullptr, nullptr, stream);

  // --- kNN graphs + encoders ---
  softmax_norm<<<NROW, 128, 0, stream>>>(l_s, Ss, wns);
  softmax_norm<<<NROW, 128, 0, stream>>>(l_t, St, wnt);
  knn_topk<<<NROW, 256, 0, stream>>>(Ss, wns, idxs);
  knn_topk<<<NROW, 256, 0, stream>>>(St, wnt, idxt);
  build_hcat<<<(NROW * 2 * FEAT + 255) / 256, 256, 0, stream>>>(fes_f, idxs, hcs);
  build_hcat<<<(NROW * 2 * FEAT + 255) / 256, 256, 0, stream>>>(fet_f, idxt, hct);
  run_linear(hcs, wtgs, b_gs, cpart, ztmp, fgs_f, fgs_b, 2 * FEAT, stream);
  run_linear(hct, wtgt, b_gt, cpart, ztmp, fgt_f, fgt_b, 2 * FEAT, stream);

  // --- loss_ge = ot(f_gt, f_es); loss_eg = ot(f_et, f_gt) ---
  run_ot_loss(fgt_b, fes_b, Mbuf, P0buf, u, v, pmin, pmax, gmm, rowSq,
              out + 3, nullptr, nullptr, stream);
  run_ot_loss(fet_b, fgt_b, Mbuf, P0buf, u, v, pmin, pmax, gmm, rowSq,
              out + 4, nullptr, nullptr, stream);

  // --- loss_g = ot(f_gt, f_gs): last call supplies output P and M ---
  float* P_out = out + 5;
  float* M_out = out + 5 + (size_t)NROW * NROW;
  run_ot_loss(fgt_b, fgs_b, Mbuf, P0buf, u, v, pmin, pmax, gmm, rowSq,
              out + 2, P_out, M_out, stream);

  total_loss<<<1, 32, 0, stream>>>(out);
}
